// GATr_4105988735452
// MI455X (gfx1250) — matrix operand order repaired and checked by reading, not compiled, not run on hardware
//
#include <hip/hip_runtime.h>
#include <math.h>

typedef __bf16 bf16;
typedef __attribute__((ext_vector_type(16))) __bf16 v16bf;
typedef __attribute__((ext_vector_type(8)))  float  v8f;

#define NT    2048
#define DIM   320      // 16 mv channels * 16 blades + 64 scalars
#define NHEAD 8
#define KTILES (DIM / 32)   // 10
#define NTILES (DIM / 16)   // 20

__constant__ int ORDER_C[16] = {0,1,2,4,8,3,5,9,6,10,12,7,11,13,14,15};

__device__ __forceinline__ float blade_sign(int a, int b) {
  int s = 0, t = a >> 1;
  while (t) { s += __popc(t & b); t >>= 1; }
  return (s & 1) ? -1.f : 1.f;
}

__device__ __forceinline__ v8f wmma_bf16(v16bf a, v16bf b, v8f c) {
  return __builtin_amdgcn_wmma_f32_16x16x32_bf16(false, a, false, b, (short)0, c,
                                                 false, false);
}

// Assemble a 16-element bf16 fragment from two contiguous 16B runs.
__device__ __forceinline__ v16bf load_frag2(const bf16* p0, const bf16* p1) {
  union { uint4 u[2]; v16bf v; } t;
  t.u[0] = *(const uint4*)p0;
  t.u[1] = *(const uint4*)p1;
  return t.v;
}

__device__ __forceinline__ float gelu_exact(float x) {
  return 0.5f * x * (1.f + erff(x * 0.70710678118654752f));
}

// ---------------------------------------------------------------------------
// Build GP / JOIN Cayley tables and the 9-dim equivariant basis on device.
// ---------------------------------------------------------------------------
__global__ void init_tables_kernel(float* gp, float* join, float* basis) {
  if (threadIdx.x != 0 || blockIdx.x != 0) return;
  int idx[16];
  for (int i = 0; i < 16; ++i) idx[ORDER_C[i]] = i;
  for (int i = 0; i < 4096; ++i) { gp[i] = 0.f; join[i] = 0.f; }
  for (int i = 0; i < 2304; ++i) basis[i] = 0.f;

  for (int i = 0; i < 16; ++i) {
    for (int j = 0; j < 16; ++j) {
      int a = ORDER_C[i], b = ORDER_C[j];
      if (!(a & b & 1))
        gp[(i * 16 + j) * 16 + idx[a ^ b]] = blade_sign(a, b);
      int da = 15 ^ a, db = 15 ^ b;
      if (!(da & db)) {
        int kb = 15 ^ (da ^ db);
        int k  = idx[kb];
        float si = blade_sign(a, da);
        float sj = blade_sign(b, db);
        float sk = blade_sign(kb, 15 ^ kb);
        join[(i * 16 + j) * 16 + k] = si * sj * sk * blade_sign(da, db);
      }
    }
  }
  for (int g = 0; g < 5; ++g)
    for (int k = 0; k < 16; ++k)
      if (__popc(ORDER_C[k]) == g) basis[(g * 16 + k) * 16 + k] = 1.f;
  for (int g = 0; g < 4; ++g)
    for (int k = 0; k < 16; ++k)
      for (int j = 0; j < 16; ++j)
        if (__popc(ORDER_C[j]) == g)
          basis[((5 + g) * 16 + k) * 16 + j] = gp[(1 * 16 + j) * 16 + k];
}

// ref = mean over (n, in_mv) of mv_in[..., 14]
__global__ void ref_kernel(const float* __restrict__ mv_in, float* refOut) {
  __shared__ float red[256];
  float s = 0.f;
  for (int i = threadIdx.x; i < NT * 3; i += 256) s += mv_in[(size_t)i * 16 + 14];
  red[threadIdx.x] = s;
  __syncthreads();
  for (int st = 128; st >= 1; st >>= 1) {
    if ((int)threadIdx.x < st) red[threadIdx.x] += red[threadIdx.x + st];
    __syncthreads();
  }
  if (threadIdx.x == 0) *refOut = red[0] / (float)(NT * 3);
}

// ---------------------------------------------------------------------------
// Fold one equi_linear layer into a 320x320 bf16 matrix, stored in B-fragment
// swizzled order: Wsw[kt][tn][lane][e], lane = half*16 + (n&15), K = e+16*half.
// Each lane's 16 elements are a contiguous 32B run -> 2x global_load_b128.
// grid = 320 blocks (k rows), 320 threads (n cols).
// ---------------------------------------------------------------------------
__global__ void prep_weights_kernel(const float* __restrict__ w_mv,
                                    const float* __restrict__ w_s2m,
                                    const float* __restrict__ w_ss,
                                    const float* __restrict__ w_m2s,
                                    const float* __restrict__ b_mv,
                                    const float* __restrict__ b_s,
                                    int im, int om, int si, int so,
                                    const float* __restrict__ basis,
                                    bf16* __restrict__ Wsw,
                                    float* __restrict__ bias) {
  int k = blockIdx.x, n = threadIdx.x;
  float v = 0.f;
  if (k < 256 && n < 256) {
    int i = k >> 4, j = k & 15, o = n >> 4, kk = n & 15;
    if (i < im && o < om) {
      float acc = 0.f;
      for (int w = 0; w < 9; ++w)
        acc += w_mv[(size_t)(o * im + i) * 9 + w] * basis[(w * 16 + kk) * 16 + j];
      v = acc;
    }
  } else if (k >= 256 && n < 256) {
    int sii = k - 256, o = n >> 4, kk = n & 15;
    if (kk == 0 && sii < si && o < om) v = w_s2m[o * si + sii];
  } else if (k < 256 && n >= 256) {
    int i = k >> 4, j = k & 15, soo = n - 256;
    if (j == 0 && i < im && soo < so) v = w_m2s[soo * im + i];
  } else {
    int sii = k - 256, soo = n - 256;
    if (sii < si && soo < so) v = w_ss[soo * si + sii];
  }
  int kt = k >> 5, rem = k & 31, half = rem >> 4, e = rem & 15;
  int tn = n >> 4, lane = half * 16 + (n & 15);
  Wsw[(((size_t)kt * NTILES + tn) * 32 + lane) * 16 + e] = (bf16)v;
  if (k == 0) {
    float bv = 0.f;
    if (n < 256) { int o = n >> 4, kk = n & 15; if (kk == 0 && o < om) bv = b_mv[o]; }
    else         { int soo = n - 256;           if (soo < so)          bv = b_s[soo]; }
    bias[n] = bv;
  }
}

// ---------------------------------------------------------------------------
// GEMM: out[m,n] = sum_k X[m,k]*W[k,n] + bias[n] (+ res).  No LDS: A and B
// fragments are direct contiguous global b128 pairs (B pre-swizzled).
// One wave32 computes a 16x64 strip (4 n-tiles) so each A fragment feeds 4
// WMMAs.  grid = (DIM/64, NT/16), block = 32.
// ---------------------------------------------------------------------------
__global__ __launch_bounds__(32)
void gemm_kernel(const bf16* __restrict__ X, const bf16* __restrict__ Wsw,
                 const float* __restrict__ bias, const float* __restrict__ res,
                 float* __restrict__ outF, int addRes) {
  int tg = blockIdx.x;                 // 4-tile group: n tiles 4*tg .. 4*tg+3
  int tm = blockIdx.y;
  int lane = threadIdx.x, half = lane >> 4, l16 = lane & 15;

  v8f acc[4] = {{}, {}, {}, {}};
  const bf16* xrow = X + (size_t)(tm * 16 + l16) * DIM;

  for (int kt = 0; kt < KTILES; ++kt) {
    // A fragment: lane row m=l16, run K = 16*half+{0..15}
    v16bf a = load_frag2(xrow + kt * 32 + 16 * half,
                         xrow + kt * 32 + 16 * half + 8);
#pragma unroll
    for (int t = 0; t < 4; ++t) {
      int tn = tg * 4 + t;
      const bf16* bp = Wsw + (((size_t)kt * NTILES + tn) * 32 + lane) * 16;
      v16bf b = load_frag2(bp, bp + 8);
      acc[t] = wmma_bf16(a, b, acc[t]);
    }
  }
#pragma unroll
  for (int t = 0; t < 4; ++t) {
#pragma unroll
    for (int r = 0; r < 8; ++r) {     // C layout: lane col l16, VGPR r -> row r+8*half
      int row = tm * 16 + r + 8 * half;
      int col = (tg * 4 + t) * 16 + l16;
      float y = acc[t][r] + bias[col];
      if (addRes) y += res[(size_t)row * DIM + col];
      outF[(size_t)row * DIM + col] = y;
    }
  }
}

// ---------------------------------------------------------------------------
// equi_ln: normalize state -> bf16 activations. One wave per token.
// ---------------------------------------------------------------------------
__global__ __launch_bounds__(32)
void equi_ln_kernel(const float* __restrict__ state, bf16* __restrict__ xout) {
  int n = blockIdx.x, lane = threadIdx.x;
  const float* p = state + (size_t)n * DIM;
  float smv = 0.f, ssc = 0.f;
  for (int i = lane; i < 256; i += 32) {
    float w = (ORDER_C[i & 15] & 1) ? 0.f : 1.f;
    float v = p[i];
    smv += v * v * w;
  }
  for (int i = lane; i < 64; i += 32) { float v = p[256 + i]; ssc += v * v; }
  for (int m = 16; m >= 1; m >>= 1) {
    smv += __shfl_xor(smv, m, 32);
    ssc += __shfl_xor(ssc, m, 32);
  }
  float inv_mv = rsqrtf(smv / 16.f + 1e-6f);
  float inv_s  = rsqrtf(ssc / 64.f + 1e-6f);
  for (int i = lane; i < 256; i += 32) xout[(size_t)n * DIM + i]       = (bf16)(p[i] * inv_mv);
  for (int i = lane; i < 64;  i += 32) xout[(size_t)n * DIM + 256 + i] = (bf16)(p[256 + i] * inv_s);
}

// Pack embed input [mv(3x16) | pad | s(4) | pad] -> bf16 [NT][DIM]
__global__ void embed_pack_kernel(const float* __restrict__ mv,
                                  const float* __restrict__ s,
                                  bf16* __restrict__ x) {
  int n = blockIdx.x, t = threadIdx.x;
  float v = 0.f;
  if (t < 48) v = mv[(size_t)n * 48 + t];
  else if (t >= 256 && t < 260) v = s[(size_t)n * 4 + (t - 256)];
  x[(size_t)n * DIM + t] = (bf16)v;
}

// q/k -> per-head padded bf16 [8][NT][64] (k masked by invariant inner product);
// v  -> TRANSPOSED bf16 [8][48][NT] so P*V B-fragments are contiguous runs.
__global__ void pack_qkv_kernel(const float* __restrict__ q, const float* __restrict__ k,
                                const float* __restrict__ v,
                                bf16* __restrict__ qf, bf16* __restrict__ kf,
                                bf16* __restrict__ vfT) {
  int n = blockIdx.x, d = threadIdx.x;   // 64 threads
  for (int h = 0; h < NHEAD; ++h) {
    float qv = 0.f, kv = 0.f, vv = 0.f;
    if (d < 32) {
      int c = 2 * h + (d >> 4), j = d & 15;
      size_t idx = (size_t)n * DIM + c * 16 + j;
      qv = q[idx]; vv = v[idx];
      kv = (ORDER_C[j] & 1) ? 0.f : k[idx];
    } else if (d < 40) {
      size_t idx = (size_t)n * DIM + 256 + h * 8 + (d - 32);
      qv = q[idx]; kv = k[idx]; vv = v[idx];
    }
    size_t o = ((size_t)h * NT + n) * 64 + d;
    qf[o] = (bf16)qv;
    kf[o] = (bf16)kv;
    if (d < 48) vfT[((size_t)h * 48 + d) * NT + n] = (bf16)vv;  // rows 40..47 zero
  }
}

// ---------------------------------------------------------------------------
// Flash-style attention: one wave owns a 16-query tile of one head.
// QK^T: WMMA over D=64 (2 K-steps).  Online softmax (16-lane shfl reductions).
// P transposed through LDS into an A-fragment; P*V from transposed V with
// contiguous b128 B-fragments.
// ---------------------------------------------------------------------------
__global__ __launch_bounds__(32)
void attn_kernel(const bf16* __restrict__ qf, const bf16* __restrict__ kf,
                 const bf16* __restrict__ vfT, bf16* __restrict__ xout) {
  int qt = blockIdx.x, h = blockIdx.y;
  int lane = threadIdx.x, half = lane >> 4, l16 = lane & 15;
  const bf16* Q  = qf  + (size_t)h * NT * 64;
  const bf16* Kp = kf  + (size_t)h * NT * 64;
  const bf16* Vt = vfT + (size_t)h * 48 * NT;

  __shared__ bf16 sP[16][40];   // P: 16 queries x 32 keys

  v16bf aq[2];
  {
    const bf16* qrow = Q + (size_t)(qt * 16 + l16) * 64;
#pragma unroll
    for (int ks = 0; ks < 2; ++ks)
      aq[ks] = load_frag2(qrow + ks * 32 + 16 * half,
                          qrow + ks * 32 + 16 * half + 8);
  }

  float runmax[8], runsum[8];
#pragma unroll
  for (int r = 0; r < 8; ++r) { runmax[r] = -1e30f; runsum[r] = 0.f; }
  v8f o0 = {}, o1 = {}, o2 = {};
  const float scale = 0.20412414523193150f;   // 1/sqrt(8*mc + sc) = 1/sqrt(24)

  for (int kc = 0; kc < NT / 32; ++kc) {
    int kb = kc * 32;

    // logits: S0 = keys kb..kb+15, S1 = keys kb+16..kb+31
    v8f s0 = {}, s1 = {};
#pragma unroll
    for (int ks = 0; ks < 2; ++ks) {
      const bf16* k0row = Kp + (size_t)(kb + l16) * 64      + ks * 32 + 16 * half;
      const bf16* k1row = Kp + (size_t)(kb + 16 + l16) * 64 + ks * 32 + 16 * half;
      v16bf b0 = load_frag2(k0row, k0row + 8);
      v16bf b1 = load_frag2(k1row, k1row + 8);
      s0 = wmma_bf16(aq[ks], b0, s0);
      s1 = wmma_bf16(aq[ks], b1, s1);
    }

    // online softmax: row m = r + 8*half lives in one 16-lane half
#pragma unroll
    for (int r = 0; r < 8; ++r) {
      float v0 = s0[r] * scale, v1 = s1[r] * scale;
      float cm = fmaxf(v0, v1);
      for (int m = 8; m >= 1; m >>= 1) cm = fmaxf(cm, __shfl_xor(cm, m, 32));
      float nm = fmaxf(runmax[r], cm);
      float cf = __expf(runmax[r] - nm);
      runmax[r] = nm;
      float p0 = __expf(v0 - nm), p1 = __expf(v1 - nm);
      s0[r] = p0; s1[r] = p1;
      float rs = p0 + p1;
      for (int m = 8; m >= 1; m >>= 1) rs += __shfl_xor(rs, m, 32);
      runsum[r] = runsum[r] * cf + rs;
      o0[r] *= cf; o1[r] *= cf; o2[r] *= cf;
    }

    // transpose P through LDS into an A-fragment (16 x 32 keys)
#pragma unroll
    for (int r = 0; r < 8; ++r) {
      sP[r + 8 * half][l16]      = (bf16)s0[r];
      sP[r + 8 * half][16 + l16] = (bf16)s1[r];
    }
    __syncthreads();
    const bf16* prow = &sP[l16][0];
    v16bf ap = load_frag2(prow + 16 * half, prow + 16 * half + 8);

    // O += P * V : B-fragment = contiguous key-run of transposed V
#pragma unroll
    for (int t = 0; t < 3; ++t) {
      const bf16* vrow = Vt + (size_t)(t * 16 + l16) * NT + kb + 16 * half;
      v16bf bv = load_frag2(vrow, vrow + 8);
      if (t == 0)      o0 = wmma_bf16(ap, bv, o0);
      else if (t == 1) o1 = wmma_bf16(ap, bv, o1);
      else             o2 = wmma_bf16(ap, bv, o2);
    }
    __syncthreads();
  }

  // finalize + scatter into merged-head layout (disjoint columns per head)
#pragma unroll
  for (int r = 0; r < 8; ++r) {
    float inv = 1.f / runsum[r];
    int row = qt * 16 + r + 8 * half;
    xout[(size_t)row * DIM + (2 * h) * 16 + l16]     = (bf16)(o0[r] * inv);
    xout[(size_t)row * DIM + (2 * h + 1) * 16 + l16] = (bf16)(o1[r] * inv);
    if (l16 < 8)
      xout[(size_t)row * DIM + 256 + h * 8 + l16]    = (bf16)(o2[r] * inv);
  }
}

// ---------------------------------------------------------------------------
// GeometricBilinear: GP on channels 0..7, JOIN*ref on 8..15, scalar-gated GELU,
// scalar path gelu(ls+rs).  One 256-thread block per token.
// ---------------------------------------------------------------------------
__global__ void bilinear_kernel(const float* __restrict__ L, const float* __restrict__ R,
                                const float* __restrict__ gp, const float* __restrict__ join,
                                const float* __restrict__ refp, bf16* __restrict__ xout) {
  int n = blockIdx.x, t = threadIdx.x;     // t = c*16 + kk
  int c = t >> 4, kk = t & 15;
  __shared__ float lm[16][16], rm[16][16], bil[16][16];
  lm[c][kk] = L[(size_t)n * DIM + t];
  rm[c][kk] = R[(size_t)n * DIM + t];
  __syncthreads();
  const float* tab = (c < 8) ? gp : join;
  float acc = 0.f;
  for (int i = 0; i < 16; ++i) {
    float lv = lm[c][i];
    for (int j = 0; j < 16; ++j) acc += tab[(i * 16 + j) * 16 + kk] * lv * rm[c][j];
  }
  if (c >= 8) acc *= *refp;
  bil[c][kk] = acc;
  __syncthreads();
  float outv = acc * gelu_exact(bil[c][0]);
  xout[(size_t)n * DIM + t] = (bf16)outv;
  if (t < 64) {
    float sv = L[(size_t)n * DIM + 256 + t] + R[(size_t)n * DIM + 256 + t];
    xout[(size_t)n * DIM + 256 + t] = (bf16)gelu_exact(sv);
  }
}

__global__ void f32_to_bf16_kernel(const float* __restrict__ in, bf16* __restrict__ out,
                                   int nElems) {
  int i = blockIdx.x * 256 + threadIdx.x;
  if (i < nElems) out[i] = (bf16)in[i];
}

__global__ void extract_pack_kernel(const float* __restrict__ y, float* __restrict__ out) {
  int n = blockIdx.x, t = threadIdx.x;
  if (t < 17)
    out[(size_t)n * 17 + t] = (t < 16) ? y[(size_t)n * DIM + t] : y[(size_t)n * DIM + 256];
}

// ---------------------------------------------------------------------------
extern "C" void kernel_launch(void* const* d_in, const int* in_sizes, int n_in,
                              void* d_out, int out_size, void* d_ws, size_t ws_size,
                              hipStream_t stream) {
  (void)in_sizes; (void)n_in; (void)out_size; (void)ws_size;
  const float* mv_in = (const float*)d_in[0];
  const float* s_in  = (const float*)d_in[1];

  // Pytree flattening: mv, s, then params leaves alphabetically:
  //   blocks[0..5] (dict keys: k,left,out,proj,q,right,v), embed, extract;
  //   each linear: b_mv,b_s,w_m2s,w_mv,w_s2m,w_ss.
  auto leaf = [&](int slot, int which) -> const float* {
    int base = (slot < 42) ? (2 + slot * 6) : (slot == 42 ? 2 + 252 : 2 + 258);
    return (const float*)d_in[base + which];
  };

  char* p = (char*)d_ws;
  auto alloc = [&](size_t bytes) -> void* {
    char* r = p; p += (bytes + 255) & ~(size_t)255; return (void*)r;
  };
  bf16*  wt    = (bf16*) alloc((size_t)44 * DIM * DIM * sizeof(bf16));
  float* biasB = (float*)alloc((size_t)44 * DIM * sizeof(float));
  float* gp    = (float*)alloc(4096 * sizeof(float));
  float* join  = (float*)alloc(4096 * sizeof(float));
  float* basis = (float*)alloc(2304 * sizeof(float));
  float* refp  = (float*)alloc(256);
  float* state = (float*)alloc((size_t)NT * DIM * sizeof(float));
  bf16*  xbuf  = (bf16*) alloc((size_t)NT * DIM * sizeof(bf16));
  float* tmp0  = (float*)alloc((size_t)NT * DIM * sizeof(float));
  float* tmp1  = (float*)alloc((size_t)NT * DIM * sizeof(float));
  float* tmp2  = (float*)alloc((size_t)NT * DIM * sizeof(float));
  bf16*  qf    = (bf16*) alloc((size_t)NHEAD * NT * 64 * sizeof(bf16));
  bf16*  kfb   = (bf16*) alloc((size_t)NHEAD * NT * 64 * sizeof(bf16));
  bf16*  vfT   = (bf16*) alloc((size_t)NHEAD * 48 * NT * sizeof(bf16));

  init_tables_kernel<<<1, 1, 0, stream>>>(gp, join, basis);
  ref_kernel<<<1, 256, 0, stream>>>(mv_in, refp);

  for (int slot = 0; slot < 44; ++slot) {
    int im = 16, om = 16, si = 64, so = 64;
    if (slot == 42) { im = 3;  om = 16; si = 4;  so = 64; }  // embed
    if (slot == 43) { im = 16; om = 1;  si = 64; so = 1;  }  // extract
    prep_weights_kernel<<<DIM, DIM, 0, stream>>>(
        leaf(slot, 3), leaf(slot, 4), leaf(slot, 5), leaf(slot, 2),
        leaf(slot, 0), leaf(slot, 1), im, om, si, so, basis,
        wt + (size_t)slot * DIM * DIM, biasB + (size_t)slot * DIM);
  }

  dim3 ggrid(DIM / 64, NT / 16);
  auto W = [&](int slot) { return wt + (size_t)slot * DIM * DIM; };
  auto B = [&](int slot) { return biasB + (size_t)slot * DIM; };

  embed_pack_kernel<<<NT, DIM, 0, stream>>>(mv_in, s_in, xbuf);
  gemm_kernel<<<ggrid, 32, 0, stream>>>(xbuf, W(42), B(42), nullptr, state, 0);

  for (int b = 0; b < 6; ++b) {
    int s0 = b * 7;   // dict key order: k=0,left=1,out=2,proj=3,q=4,right=5,v=6
    // --- attention block ---
    equi_ln_kernel<<<NT, 32, 0, stream>>>(state, xbuf);
    gemm_kernel<<<ggrid, 32, 0, stream>>>(xbuf, W(s0 + 4), B(s0 + 4), nullptr, tmp0, 0); // q
    gemm_kernel<<<ggrid, 32, 0, stream>>>(xbuf, W(s0 + 0), B(s0 + 0), nullptr, tmp1, 0); // k
    gemm_kernel<<<ggrid, 32, 0, stream>>>(xbuf, W(s0 + 6), B(s0 + 6), nullptr, tmp2, 0); // v
    pack_qkv_kernel<<<NT, 64, 0, stream>>>(tmp0, tmp1, tmp2, qf, kfb, vfT);
    attn_kernel<<<dim3(NT / 16, NHEAD), 32, 0, stream>>>(qf, kfb, vfT, xbuf);
    gemm_kernel<<<ggrid, 32, 0, stream>>>(xbuf, W(s0 + 3), B(s0 + 3), state, state, 1);  // proj + residual
    // --- MLP block ---
    equi_ln_kernel<<<NT, 32, 0, stream>>>(state, xbuf);
    gemm_kernel<<<ggrid, 32, 0, stream>>>(xbuf, W(s0 + 1), B(s0 + 1), nullptr, tmp0, 0); // left
    gemm_kernel<<<ggrid, 32, 0, stream>>>(xbuf, W(s0 + 5), B(s0 + 5), nullptr, tmp1, 0); // right
    bilinear_kernel<<<NT, 256, 0, stream>>>(tmp0, tmp1, gp, join, refp, xbuf);
    gemm_kernel<<<ggrid, 32, 0, stream>>>(xbuf, W(s0 + 2), B(s0 + 2), state, state, 1);  // out + residual
  }

  f32_to_bf16_kernel<<<(NT * DIM + 255) / 256, 256, 0, stream>>>(state, xbuf, NT * DIM);
  gemm_kernel<<<ggrid, 32, 0, stream>>>(xbuf, W(43), B(43), nullptr, tmp0, 0);
  extract_pack_kernel<<<NT, 32, 0, stream>>>(tmp0, (float*)d_out);
}
